// Router_71133248356524
// MI455X (gfx1250) — matrix operand order repaired and checked by reading, compiled, not run on hardware
//
#include <hip/hip_runtime.h>
#include <hip/hip_bf16.h>

typedef __attribute__((ext_vector_type(8)))  float  v8f;
typedef __attribute__((ext_vector_type(16))) __bf16 v16bf;
typedef __attribute__((ext_vector_type(4)))  int    v4i;

#define N_TOK    16384
#define D_MODEL  2048
#define N_EXP    64
#define CAPACITY 640      /* int(1.25 * 16384 * 2 / 64) */
#define KC       256      /* K-chunk staged in LDS */
#define NC       (D_MODEL / KC)

#if __has_builtin(__builtin_amdgcn_global_load_async_to_lds_b128)
#define HAVE_ASYNC_LDS 1
typedef __attribute__((address_space(1))) v4i* gptr_v4i;
typedef __attribute__((address_space(3))) v4i* lptr_v4i;
#else
#define HAVE_ASYNC_LDS 0
#endif

// ---------------------------------------------------------------- utilities
__device__ __forceinline__ float wred_sum(float v) {
#pragma unroll
    for (int m = 16; m > 0; m >>= 1) v += __shfl_xor(v, m, 32);
    return v;
}
__device__ __forceinline__ float wred_max(float v) {
#pragma unroll
    for (int m = 16; m > 0; m >>= 1) v = fmaxf(v, __shfl_xor(v, m, 32));
    return v;
}

// ---------------------------------------------------------- K0: W fp32->bf16
__global__ __launch_bounds__(256) void moe_cvt_w_kernel(const float* __restrict__ W,
                                                        __bf16* __restrict__ Wbf) {
    int i = blockIdx.x * 256 + threadIdx.x;
    if (i < N_EXP * D_MODEL) Wbf[i] = (__bf16)W[i];
}

// ------------------------------------------------------- K0b: zero workspace
__global__ __launch_bounds__(64) void moe_init_kernel(int* counts0, int* countsF,
                                                      float* cnorm, float* zacc) {
    int i = threadIdx.x;
    if (i < N_EXP) { counts0[i] = 0; countsF[i] = 0; cnorm[i] = 0.0f; }
    if (i == 0) *zacc = 0.0f;
}

// ------------------------------------------------- K1: logits GEMM via WMMA
// 256 threads = 8 waves; each wave owns a 16-token x 64-expert C tile.
// W (bf16) is staged in LDS in double-buffered K-chunks via gfx1250 async
// LDS loads (ASYNCcnt), so B fragments come from ds_load_b128 instead of L2.
__global__ __launch_bounds__(256) void moe_gemm_kernel(const float* __restrict__ X,
                                                       const __bf16* __restrict__ Wbf,
                                                       float* __restrict__ logits) {
    __shared__ __bf16 shW[2][N_EXP * KC];   // 2 x 32 KB

    const int lane = threadIdx.x & 31;
    const int wave = threadIdx.x >> 5;
    const int row_base = (blockIdx.x * 8 + wave) * 16;
    const int m   = lane & 15;
    const int grp = lane >> 4;

    // cooperative staging: thread t copies one quarter-row (64 bf16 = 128 B)
    const int s_row  = threadIdx.x >> 2;          // expert row 0..63
    const int s_quad = (threadIdx.x & 3) * 64;    // element offset in chunk row
    const __bf16* sg_base = Wbf + (size_t)s_row * D_MODEL + s_quad;
    __bf16*       sl_base = &shW[0][0] + s_row * KC + s_quad;

    const float* arow = X + (size_t)(row_base + m) * D_MODEL;

    v8f c0 = {}, c1 = {}, c2 = {}, c3 = {};

    // ---- stage chunk 0 ----
#if HAVE_ASYNC_LDS
#pragma unroll
    for (int i = 0; i < 8; ++i)
        __builtin_amdgcn_global_load_async_to_lds_b128(
            (gptr_v4i)(sg_base + i * 8), (lptr_v4i)(sl_base + i * 8), 0, 0);
#else
#pragma unroll
    for (int i = 0; i < 8; ++i)
        *(uint4*)(sl_base + i * 8) = *(const uint4*)(sg_base + i * 8);
#endif

    for (int c = 0; c < NC; ++c) {
        // ---- issue next chunk into the other buffer ----
        if (c + 1 < NC) {
            const __bf16* g = sg_base + (c + 1) * KC;
            __bf16*       l = &shW[(c + 1) & 1][0] + s_row * KC + s_quad;
#if HAVE_ASYNC_LDS
#pragma unroll
            for (int i = 0; i < 8; ++i)
                __builtin_amdgcn_global_load_async_to_lds_b128(
                    (gptr_v4i)(g + i * 8), (lptr_v4i)(l + i * 8), 0, 0);
            __builtin_amdgcn_s_wait_asynccnt(8);   // chunk c complete (in-order)
#else
#pragma unroll
            for (int i = 0; i < 8; ++i)
                *(uint4*)(l + i * 8) = *(const uint4*)(g + i * 8);
#endif
        } else {
#if HAVE_ASYNC_LDS
            __builtin_amdgcn_s_wait_asynccnt(0);
#endif
        }
        __syncthreads();   // chunk c visible to all waves

        const __bf16* buf = &shW[c & 1][0];
        const int ck = c * KC;
        __builtin_prefetch(arow + ck + KC, 0, 1);  // stream-ahead on HBM-bound A

#pragma unroll
        for (int kk = 0; kk < KC; kk += 32) {
            // A fragment: group0 -> K {0..15}, group1 -> K {16..31}
            const int ka = ck + kk + grp * 16;
            const int kb = ck + kk + 8 + grp * 16;
            const float4 a0 = *(const float4*)(arow + ka);
            const float4 a1 = *(const float4*)(arow + ka + 4);
            const float4 a2 = *(const float4*)(arow + kb);
            const float4 a3 = *(const float4*)(arow + kb + 4);
            v16bf a;
            a[0]  = (__bf16)a0.x; a[1]  = (__bf16)a0.y; a[2]  = (__bf16)a0.z; a[3]  = (__bf16)a0.w;
            a[4]  = (__bf16)a1.x; a[5]  = (__bf16)a1.y; a[6]  = (__bf16)a1.z; a[7]  = (__bf16)a1.w;
            a[8]  = (__bf16)a2.x; a[9]  = (__bf16)a2.y; a[10] = (__bf16)a2.z; a[11] = (__bf16)a2.w;
            a[12] = (__bf16)a3.x; a[13] = (__bf16)a3.y; a[14] = (__bf16)a3.z; a[15] = (__bf16)a3.w;

            // B fragments from LDS: lane group holds 16 contiguous K (32 B)
            const int kbb = kk + grp * 16;
            v16bf b0 = *(const v16bf*)(buf + (m)      * KC + kbb);
            v16bf b1 = *(const v16bf*)(buf + (m + 16) * KC + kbb);
            v16bf b2 = *(const v16bf*)(buf + (m + 32) * KC + kbb);
            v16bf b3 = *(const v16bf*)(buf + (m + 48) * KC + kbb);

            c0 = __builtin_amdgcn_wmma_f32_16x16x32_bf16(false, a, false, b0, (short)0, c0, false, false);
            c1 = __builtin_amdgcn_wmma_f32_16x16x32_bf16(false, a, false, b1, (short)0, c1, false, false);
            c2 = __builtin_amdgcn_wmma_f32_16x16x32_bf16(false, a, false, b2, (short)0, c2, false, false);
            c3 = __builtin_amdgcn_wmma_f32_16x16x32_bf16(false, a, false, b3, (short)0, c3, false, false);
        }
        __syncthreads();   // all waves done with buffer before it is re-staged
    }

    // C layout: VGPR r -> row grp*8 + r, col = lane&15 (+16 per tile)
#pragma unroll
    for (int r = 0; r < 8; ++r) {
        const int row = row_base + grp * 8 + r;
        float* lr = logits + (size_t)row * N_EXP + m;
        lr[0]  = c0[r];
        lr[16] = c1[r];
        lr[32] = c2[r];
        lr[48] = c3[r];
    }
}

// -------------------------- K2: softmax + top-2 + round-0 histogram + z-loss
__global__ __launch_bounds__(256) void moe_softmax_top2_kernel(float* __restrict__ rw,
                                                               int* __restrict__ top1,
                                                               int* __restrict__ top2,
                                                               float* __restrict__ w1o,
                                                               float* __restrict__ w2o,
                                                               int* __restrict__ counts0,
                                                               float* __restrict__ zacc) {
    const int lane = threadIdx.x & 31;
    const int wv   = threadIdx.x >> 5;
    const int t    = blockIdx.x * 8 + wv;

    float* row = rw + (size_t)t * N_EXP;
    const float l0 = row[lane];
    const float l1 = row[lane + 32];

    float zs = wred_sum(l0 * l0 + l1 * l1);
    if (lane == 0) atomicAdd(zacc, zs);

    const float mx = wred_max(fmaxf(l0, l1));
    const float p0 = __expf(l0 - mx);
    const float p1 = __expf(l1 - mx);
    const float inv = 1.0f / wred_sum(p0 + p1);
    const float r0 = p0 * inv;
    const float r1 = p1 * inv;
    row[lane]      = r0;   // routing_weights output (in-place over logits)
    row[lane + 32] = r1;

    // top-2 (ties -> lower index, matching lax.top_k)
    float w1v, w2v; int i1, i2;
    if (r1 > r0) { w1v = r1; i1 = lane + 32; w2v = r0; i2 = lane; }
    else         { w1v = r0; i1 = lane;      w2v = r1; i2 = lane + 32; }
#pragma unroll
    for (int mk = 16; mk > 0; mk >>= 1) {
        float ow1 = __shfl_xor(w1v, mk, 32); int oi1 = __shfl_xor(i1, mk, 32);
        float ow2 = __shfl_xor(w2v, mk, 32); int oi2 = __shfl_xor(i2, mk, 32);
        bool ofirst = (ow1 > w1v) || (ow1 == w1v && oi1 < i1);
        if (ofirst) {
            bool keep = (w1v > ow2) || (w1v == ow2 && i1 < oi2);
            if (keep) { w2v = w1v; i2 = i1; } else { w2v = ow2; i2 = oi2; }
            w1v = ow1; i1 = oi1;
        } else {
            bool rep = (ow1 > w2v) || (ow1 == w2v && oi1 < i2);
            if (rep) { w2v = ow1; i2 = oi1; }
        }
    }

    if (lane == 0) {
        const float sw = w1v + w2v + 1e-8f;
        top1[t] = i1;
        top2[t] = i2;
        w1o[t]  = w1v / sw;
        w2o[t]  = w2v / sw;
        atomicAdd(&counts0[i1], 1);   // round 0: all top-1 admitted (cap 640 > 0)
    }
}

// ------------------------------------- K3: round-1 capacity + final histogram
__global__ __launch_bounds__(256) void moe_round1_kernel(const int* __restrict__ top1,
                                                         const int* __restrict__ top2,
                                                         const int* __restrict__ counts0,
                                                         int* __restrict__ countsF,
                                                         int* __restrict__ valid2) {
    const int t  = blockIdx.x * 256 + threadIdx.x;
    const int t2 = top2[t];
    const int v  = counts0[t2] < CAPACITY ? 1 : 0;
    valid2[t] = v;
    atomicAdd(&countsF[top1[t]], 1);
    if (v) atomicAdd(&countsF[t2], 1);
}

// ----------------------------- K4a: least-loaded expert (argmin final counts)
__global__ __launch_bounds__(32) void moe_argmin_kernel(const int* __restrict__ countsF,
                                                        int* __restrict__ least) {
    const int lane = threadIdx.x;
    int c0 = countsF[lane], c1 = countsF[lane + 32];
    int bi = (c1 < c0) ? lane + 32 : lane;
    int bc = min(c0, c1);
#pragma unroll
    for (int mk = 16; mk > 0; mk >>= 1) {
        int oc = __shfl_xor(bc, mk, 32);
        int oi = __shfl_xor(bi, mk, 32);
        if (oc < bc || (oc == bc && oi < bi)) { bc = oc; bi = oi; }
    }
    if (lane == 0) *least = bi;
}

// -------------------------- K4b: normalized dispatch mask + per-expert sums
__global__ __launch_bounds__(256) void moe_dispatch_kernel(const int* __restrict__ top1,
                                                           const int* __restrict__ top2,
                                                           const float* __restrict__ w1a,
                                                           const float* __restrict__ w2a,
                                                           const int* __restrict__ valid2,
                                                           const int* __restrict__ least,
                                                           float* __restrict__ mask,
                                                           float* __restrict__ cnorm) {
    const int lane = threadIdx.x & 31;
    const int wv   = threadIdx.x >> 5;
    const int t    = blockIdx.x * 8 + wv;

    const int   t1 = top1[t], t2 = top2[t];
    const float a  = w1a[t];
    const float b  = valid2[t] ? w2a[t] : 0.0f;
    const float rowsum = a + b;
    const bool  unrouted = (rowsum == 0.0f);
    const int   L = *least;
    const float denom = (unrouted ? 1.0f : rowsum) + 1e-8f;

    float* mrow = mask + (size_t)t * N_EXP;
#pragma unroll
    for (int h = 0; h < 2; ++h) {
        const int e = lane + 32 * h;
        float v = unrouted ? (e == L ? 1.0f : 0.0f)
                           : ((e == t1 ? a : 0.0f) + (e == t2 ? b : 0.0f));
        const float o = v / denom;
        mrow[e] = o;
        if (o != 0.0f) atomicAdd(&cnorm[e], o);
    }
}

// ----------------------------------------------------- K5: router loss scalar
__global__ __launch_bounds__(32) void moe_loss_kernel(const float* __restrict__ cnorm,
                                                      const float* __restrict__ zacc,
                                                      float* __restrict__ loss) {
    const int lane = threadIdx.x;
    const float invN   = 1.0f / (float)N_TOK;
    const float target = (float)(N_TOK * 2 / N_EXP) * invN;   // 512/16384
    float d0 = cnorm[lane] * invN - target;
    float d1 = cnorm[lane + 32] * invN - target;
    float s = wred_sum(d0 * d0 + d1 * d1);
    if (lane == 0) {
        float lb = s / (float)N_EXP;
        float z  = (*zacc) / (float)(N_TOK * N_EXP);
        loss[0] = 0.001f * z + 0.001f * lb;
    }
}

// ---------------------------------------------------------------------------
extern "C" void kernel_launch(void* const* d_in, const int* in_sizes, int n_in,
                              void* d_out, int out_size, void* d_ws, size_t ws_size,
                              hipStream_t stream) {
    const float* x = (const float*)d_in[0];   // (4,4096,2048) f32
    const float* W = (const float*)d_in[1];   // (64,2048) f32

    float* out  = (float*)d_out;
    float* rw   = out;                              // routing_weights region
    float* mask = out + (size_t)N_TOK * N_EXP;      // dispatch mask region
    float* loss = out + (size_t)2 * N_TOK * N_EXP;  // scalar

    char* ws = (char*)d_ws;
    __bf16* Wbf    = (__bf16*)(ws);                           // 256 KB
    int*   top1    = (int*)  (ws + 262144);                   // 64 KB each
    int*   top2    = (int*)  (ws + 262144 + 1 * 65536);
    float* w1a     = (float*)(ws + 262144 + 2 * 65536);
    float* w2a     = (float*)(ws + 262144 + 3 * 65536);
    int*   valid2  = (int*)  (ws + 262144 + 4 * 65536);
    int*   counts0 = (int*)  (ws + 262144 + 5 * 65536);
    int*   countsF = (int*)  (ws + 262144 + 5 * 65536 + 256);
    float* cnorm   = (float*)(ws + 262144 + 5 * 65536 + 512);
    float* zacc    = (float*)(ws + 262144 + 5 * 65536 + 768);
    int*   least   = (int*)  (ws + 262144 + 5 * 65536 + 772);

    moe_cvt_w_kernel<<<(N_EXP * D_MODEL + 255) / 256, 256, 0, stream>>>(W, Wbf);
    moe_init_kernel<<<1, 64, 0, stream>>>(counts0, countsF, cnorm, zacc);
    moe_gemm_kernel<<<N_TOK / 128, 256, 0, stream>>>(x, Wbf, rw);
    moe_softmax_top2_kernel<<<N_TOK / 8, 256, 0, stream>>>(rw, top1, top2, w1a, w2a,
                                                           counts0, zacc);
    moe_round1_kernel<<<N_TOK / 256, 256, 0, stream>>>(top1, top2, counts0, countsF, valid2);
    moe_argmin_kernel<<<1, 32, 0, stream>>>(countsF, least);
    moe_dispatch_kernel<<<N_TOK / 8, 256, 0, stream>>>(top1, top2, w1a, w2a, valid2,
                                                       least, mask, cnorm);
    moe_loss_kernel<<<1, 32, 0, stream>>>(cnorm, zacc, loss);
}
